// AllegroQeq_54674933678519
// MI455X (gfx1250) — matrix operand order repaired and checked by reading, not compiled, not run on hardware
//
#include <hip/hip_runtime.h>
#include <cmath>

#define N_ATOMS 20000
#define DEG     16
#define E_EDGES (N_ATOMS * DEG)
#define S_SPEC  100
#define EMBD    32
#define NRBF    8
#define HDIM    128
#define MULD    16

typedef __attribute__((ext_vector_type(16))) _Float16 v16h;
typedef __attribute__((ext_vector_type(8)))  float    v8f;

__device__ __forceinline__ float frcp(float x)     { return __builtin_amdgcn_rcpf(x); }
__device__ __forceinline__ float fsilu(float x)    { return x * frcp(1.f + __expf(-x)); }
__device__ __forceinline__ float fsigmoid(float x) { return frcp(1.f + __expf(-x)); }
__device__ __forceinline__ float fsoftplus(float x){ return log1pf(__expf(x)); }

__device__ __forceinline__ void atomAddF(float* p, float v) {
#if defined(__AMDGCN__)
    unsafeAtomicAdd(p, v);     // native global_atomic_add_f32
#else
    atomicAdd(p, v);
#endif
}

// ---- WMMA operand layouts (CDNA5 ISA 7.12.2, wave32) ----
// A (16x32 f16): lane ln(0..15) row M=ln holds K = {0..7,16..23}; lane ln+16 holds {8..15,24..31}.
// We stage A rows in LDS with K contiguous (lane half h holds K=16h..16h+15, matching B).
__device__ __forceinline__ int a_idx(int m, int k) {
    int kt = k >> 5, kk = k & 31;
    return ((kt << 4) + m) * 32 + kk;
}
// packed B (32x16 f16): lane ln holds column N=ln, K=0..15; lane ln+16 holds K=16..31 (contiguous halves).
__device__ __forceinline__ size_t b_idx(int k, int n, int ntiles) {
    int kt = k >> 5, kk = k & 31, nt = n >> 4, nn = n & 15;
    return ((size_t)(kt * ntiles + nt) * 16 + nn) * 32 + kk;
}

// One wave computes a 16(edge) x (ntiles*16) GEMM: lA (LDS, A-layout, Kpad=ktiles*32)
// times packed gB. Optional SiLU, per-row scale, and multiple output destinations.
__device__ __forceinline__ void wave_gemm16(
    const _Float16* lA, const _Float16* gB, int ktiles, int ntiles,
    bool do_silu, const float* lEnv,
    _Float16* lOutA,            // LDS dest, A-layout (for chaining)
    float* lOutRM,              // LDS dest, row-major f32 [16][ntiles*16]
    _Float16* gOutH, int ghdim, // global f16 dest, row-major
    float* gOutF, int gfdim,    // global f32 dest, row-major
    int lane)
{
    const int half = lane >> 4, ln = lane & 15;
    for (int nt = 0; nt < ntiles; ++nt) {
        v8f acc = {};
        for (int kt = 0; kt < ktiles; ++kt) {
            v16h a = *(const v16h*)(lA + ((kt << 4) + ln) * 32 + (half << 4));
            v16h b = *(const v16h*)(gB + ((size_t)(kt * ntiles + nt) * 16 + ln) * 32 + (half << 4));
            acc = __builtin_amdgcn_wmma_f32_16x16x32_f16(
                false, a, false, b, (short)0, acc, false, false);
        }
#pragma unroll
        for (int i = 0; i < 8; ++i) {
            int row = i + (half << 3);        // C/D: VGPR i -> M=i (lanes 0-15), M=i+8 (16-31)
            int col = (nt << 4) + ln;
            float v = acc[i];
            if (do_silu) v = fsilu(v);
            if (lEnv)    v *= lEnv[row];
            if (lOutA)   lOutA[a_idx(row, col)] = (_Float16)v;
            if (lOutRM)  lOutRM[row * (ntiles << 4) + col] = v;
            if (gOutH)   gOutH[(size_t)row * ghdim + col] = (_Float16)v;
            if (gOutF)   gOutF[(size_t)row * gfdim + col] = v;
        }
    }
}

// ---- weight pre-pack: f32 row-major (K x Nout) -> f16 B-operand tiles, scale baked in ----
__global__ void pack_w(const float* __restrict__ W, _Float16* __restrict__ dst,
                       int K, int Kpad, int Nout, float scale)
{
    int idx = blockIdx.x * blockDim.x + threadIdx.x;
    int total = Kpad * Nout;
    if (idx >= total) return;
    int k = idx / Nout, n = idx % Nout;
    float v = (k < K) ? W[(size_t)k * Nout + n] * scale : 0.f;
    dst[b_idx(k, n, Nout >> 4)] = (_Float16)v;
}

// ---------------- stage 1: edge features + 2-body MLP + v/w + wY scatter ----------------
struct WS1 {
    _Float16 A0[2048];
    _Float16 A1[2048];
    float    RM[256];
    float    Yv[256];
    float    env[16];
    int      snd[16];
};

__global__ void __launch_bounds__(64) k1_edge_mlp(
    const float* __restrict__ vectors, const int* __restrict__ senders,
    const int* __restrict__ receivers, const int* __restrict__ species,
    const float* __restrict__ embed,
    const _Float16* __restrict__ pW2b0, const _Float16* __restrict__ pW2b1,
    const _Float16* __restrict__ pW2b2, const _Float16* __restrict__ pWv,
    const _Float16* __restrict__ pWl_w,
    _Float16* __restrict__ x1buf, float* __restrict__ Ybuf,
    float* __restrict__ vbuf, float* __restrict__ wYacc)
{
    __shared__ WS1 ws[2];
    const int wv = threadIdx.x >> 5, lane = threadIdx.x & 31;
    const int half = lane >> 4, ln = lane & 15;
    WS1& W = ws[wv];
    const int tile = blockIdx.x * 2 + wv;
    if (tile * 16 >= E_EDGES) return;
    const int e0 = tile * 16;

    if (half == 0) {
        int e = e0 + ln;
        float vx = vectors[e * 3 + 0], vy = vectors[e * 3 + 1], vz = vectors[e * 3 + 2];
        float r = sqrtf(vx * vx + vy * vy + vz * vz);
        float u3 = r * r * r, u6 = u3 * u3;
        float env = 1.f - 28.f * u6 + 48.f * u6 * r - 21.f * u6 * r * r;   // P=6
        env = (r < 1.f) ? env : 0.f;
        W.env[ln] = env;
        int se = senders[e];
        W.snd[ln] = se;
        float inv_r = frcp(r);
        float x = vx * inv_r, y = vy * inv_r, z = vz * inv_r;
        const float s3 = 1.7320508075688772f, s5 = 2.23606797749979f;
        const float s15 = 3.872983346207417f, s7 = 2.6457513110645907f;
        const float c35 = 2.0916500663351889f, c105 = 10.246950765959598f;
        const float c21 = 1.6201851746019651f;
        float Y[16];
        Y[0] = 1.f;          Y[1] = s3 * x;      Y[2] = s3 * y;      Y[3] = s3 * z;
        Y[4] = s15 * x * y;  Y[5] = s15 * y * z;
        Y[6] = 0.5f * s5 * (3.f * z * z - 1.f);  Y[7] = s15 * x * z;
        Y[8] = 0.5f * s15 * (x * x - y * y);
        Y[9]  = c35 * y * (3.f * x * x - y * y);
        Y[10] = c105 * x * y * z;
        Y[11] = c21 * y * (5.f * z * z - 1.f);
        Y[12] = 0.5f * s7 * z * (5.f * z * z - 3.f);
        Y[13] = c21 * x * (5.f * z * z - 1.f);
        Y[14] = 0.5f * c105 * z * (x * x - y * y);
        Y[15] = c35 * x * (x * x - 3.f * y * y);
        for (int k = 0; k < 16; ++k) {
            W.Yv[ln * 16 + k] = Y[k];
            Ybuf[(size_t)e * 16 + k] = Y[k];
        }
        // input features: [emb[sender](32), emb[receiver](32), rbf(8)], Kpad=96
        int ss = species[se];
        int sr = species[receivers[e]];
        const float SQ2 = 1.4142135623730951f, PI = 3.14159265358979323f;
        for (int k = 0; k < 32; ++k) W.A0[a_idx(ln, k)]      = (_Float16)embed[ss * EMBD + k];
        for (int k = 0; k < 32; ++k) W.A0[a_idx(ln, 32 + k)] = (_Float16)embed[sr * EMBD + k];
        for (int n = 1; n <= NRBF; ++n)
            W.A0[a_idx(ln, 63 + n)] = (_Float16)(SQ2 * __sinf(PI * n * r) * inv_r * env);
        for (int k = 72; k < 96; ++k) W.A0[a_idx(ln, k)] = (_Float16)0.f;
    }

    // 3-layer 2-body MLP: silu, silu, (none) * env
    wave_gemm16(W.A0, pW2b0, 3, 8, true,  nullptr, W.A1, nullptr, nullptr, 0, nullptr, 0, lane);
    wave_gemm16(W.A1, pW2b1, 4, 8, true,  nullptr, W.A0, nullptr, nullptr, 0, nullptr, 0, lane);
    wave_gemm16(W.A0, pW2b2, 4, 8, false, W.env,   W.A1, nullptr,
                x1buf + (size_t)e0 * HDIM, HDIM, nullptr, 0, lane);
    // v = x1 @ Wv, w = x1 @ Wl_w (no activation)
    wave_gemm16(W.A1, pWv,   4, 1, false, nullptr, nullptr, nullptr, nullptr, 0,
                vbuf + (size_t)e0 * 16, 16, lane);
    wave_gemm16(W.A1, pWl_w, 4, 1, false, nullptr, nullptr, W.RM, nullptr, 0, nullptr, 0, lane);

    // scatter w[e,m] * Y[e,k] into wY[sender] (segment_sum)
    {
        int m = ln;
#pragma unroll
        for (int i = 0; i < 8; ++i) {
            int row = i + (half << 3);
            float wval = W.RM[row * 16 + m];
            float* basep = wYacc + (size_t)W.snd[row] * 256 + m * 16;
            for (int k = 0; k < 16; ++k)
                atomAddF(basep + k, wval * W.Yv[row * 16 + k]);
        }
    }
}

// ---------------- stage 2: scal[e,m] = eps_tp * v[e,m] * sum_k wY[s[e],m,k]*Y[e,k] ----------------
__global__ void k2_scal(const int* __restrict__ senders, const float* __restrict__ wYacc,
                        const float* __restrict__ Ybuf, const float* __restrict__ vbuf,
                        const float* __restrict__ varepsilon, _Float16* __restrict__ scalbuf)
{
    int idx = blockIdx.x * blockDim.x + threadIdx.x;
    if (idx >= E_EDGES * 16) return;
    int e = idx >> 4, m = idx & 15;
    float eps_tp = rsqrtf(1.f + fsoftplus(varepsilon[0]));
    const float* wy = wYacc + (size_t)senders[e] * 256 + m * 16;
    const float* Ye = Ybuf + (size_t)e * 16;
    float acc = 0.f;
#pragma unroll
    for (int k = 0; k < 16; ++k) acc += wy[k] * Ye[k];
    scalbuf[idx] = (_Float16)(acc * vbuf[idx] * eps_tp);
}

// ---------------- stage 3: latent MLP (K=144) + chi/sig/eps heads ----------------
struct WS3 {
    _Float16 A0[2560];   // Kpad=160
    _Float16 A1[2048];
    float    hRM[512];
    float    env[16];
    int      snd[16];
};

__global__ void __launch_bounds__(64) k3_edge_mlp2(
    const float* __restrict__ vectors, const int* __restrict__ senders,
    const _Float16* __restrict__ x1buf, const _Float16* __restrict__ scalbuf,
    const _Float16* __restrict__ pWl_x0, const _Float16* __restrict__ pWl_x1,
    const _Float16* __restrict__ pWl_x2,
    const _Float16* __restrict__ pWchi0, const _Float16* __restrict__ pWsig0,
    const _Float16* __restrict__ pWeps0,
    const float* __restrict__ Wchi1, const float* __restrict__ Wsig1,
    const float* __restrict__ Weps1,
    _Float16* __restrict__ x2buf, float* __restrict__ chis_acc,
    float* __restrict__ sig_acc, float* __restrict__ eps_acc)
{
    __shared__ WS3 ws[2];
    const int wv = threadIdx.x >> 5, lane = threadIdx.x & 31;
    const int half = lane >> 4, ln = lane & 15;
    WS3& W = ws[wv];
    const int tile = blockIdx.x * 2 + wv;
    if (tile * 16 >= E_EDGES) return;
    const int e0 = tile * 16;

    if (half == 0) {
        int e = e0 + ln;
        float vx = vectors[e * 3 + 0], vy = vectors[e * 3 + 1], vz = vectors[e * 3 + 2];
        float r = sqrtf(vx * vx + vy * vy + vz * vz);
        float u6 = r * r * r; u6 *= u6;
        float env = 1.f - 28.f * u6 + 48.f * u6 * r - 21.f * u6 * r * r;
        W.env[ln] = (r < 1.f) ? env : 0.f;
        W.snd[ln] = senders[e];
    }
    {   // A = [x1 (128), scal (16), 0-pad(16)]
        int e = e0 + ln;
        const _Float16* x1 = x1buf + (size_t)e * HDIM;
        const _Float16* sc = scalbuf + (size_t)e * 16;
        int k0 = half * 80;
        for (int kk = 0; kk < 80; ++kk) {
            int k = k0 + kk;
            _Float16 v = (k < 128) ? x1[k] : (k < 144 ? sc[k - 128] : (_Float16)0.f);
            W.A0[a_idx(ln, k)] = v;
        }
    }

    wave_gemm16(W.A0, pWl_x0, 5, 8, true,  nullptr, W.A1, nullptr, nullptr, 0, nullptr, 0, lane);
    wave_gemm16(W.A1, pWl_x1, 4, 8, true,  nullptr, W.A0, nullptr, nullptr, 0, nullptr, 0, lane);
    wave_gemm16(W.A0, pWl_x2, 4, 8, false, W.env,   W.A1, nullptr,
                x2buf + (size_t)e0 * HDIM, HDIM, nullptr, 0, lane);

    // chi head: silu(x@Wchi0/sqrt128) @ Wchi1/4
    wave_gemm16(W.A1, pWchi0, 4, 1, true, nullptr, nullptr, W.hRM, nullptr, 0, nullptr, 0, lane);
    if (half == 0) {
        float acc = 0.f;
#pragma unroll
        for (int m = 0; m < 16; ++m) acc += W.hRM[ln * 16 + m] * Wchi1[m];
        atomAddF(chis_acc + W.snd[ln], acc * 0.25f);
    }
    // sig head: 128->32->1
    wave_gemm16(W.A1, pWsig0, 4, 2, true, nullptr, nullptr, W.hRM, nullptr, 0, nullptr, 0, lane);
    if (half == 0) {
        float acc = 0.f;
#pragma unroll
        for (int t = 0; t < 32; ++t) acc += W.hRM[ln * 32 + t] * Wsig1[t];
        atomAddF(sig_acc + W.snd[ln], acc * 0.17677669529663687f);
    }
    // eps head: 128->32->1
    wave_gemm16(W.A1, pWeps0, 4, 2, true, nullptr, nullptr, W.hRM, nullptr, 0, nullptr, 0, lane);
    if (half == 0) {
        float acc = 0.f;
#pragma unroll
        for (int t = 0; t < 32; ++t) acc += W.hRM[ln * 32 + t] * Weps1[t];
        atomAddF(eps_acc + W.snd[ln], acc * 0.17677669529663687f);
    }
}

// ---------------- stage 4: per-node QEq prep + sig/eps activation ----------------
// scalars: [0]=numer [1]=denom [2]=node_e [3]=pot [4]=vdw [5]=edge_e
__global__ void k4_node(const int* __restrict__ species, const float* __restrict__ radius,
                        const float* __restrict__ hardness, const float* __restrict__ pe,
                        const float* __restrict__ chis_acc, const float* __restrict__ sig_acc,
                        const float* __restrict__ eps_acc,
                        float* __restrict__ Jbuf, float* __restrict__ sig_n,
                        float* __restrict__ epn_n, float* __restrict__ scalars)
{
    int i = blockIdx.x * blockDim.x + threadIdx.x;
    if (i >= N_ATOMS) return;
    int sp = species[i];
    float gamma = radius[sp] * 4.f + 0.5f;
    float hard  = fsoftplus(hardness[sp]);
    float J = hard + 1.f / (gamma * 1.7724538509055159f);   // sqrt(pi)
    Jbuf[i] = J;
    atomAddF(&scalars[0], chis_acc[i] / J);
    atomAddF(&scalars[1], 1.f / J);
    atomAddF(&scalars[2], pe[sp]);
    sig_n[i] = fsigmoid(sig_acc[i]) * 0.15f + 0.15f;
    epn_n[i] = fsigmoid(eps_acc[i]) * 1.7f + 0.3f;
}

// ---------------- stage 5: q, pot, wq = [q, charge_embed] @ Wq0 / sqrt(17) ----------------
__global__ void k5_node_q(const int* __restrict__ species, const float* __restrict__ chis_acc,
                          const float* __restrict__ Jbuf, const float* __restrict__ charge_embed,
                          const float* __restrict__ Wq0, float* __restrict__ scalars,
                          _Float16* __restrict__ wqbuf)
{
    int i = blockIdx.x * blockDim.x + threadIdx.x;
    if (i >= N_ATOMS) return;
    float lam  = scalars[0] / scalars[1];
    float chis = chis_acc[i];
    float J    = Jbuf[i];
    float q    = (lam - chis) / J;
    atomAddF(&scalars[3], chis * q + 0.5f * J * q * q);
    int sp = species[i];
    const float is17 = 0.24253562503633297f;  // 1/sqrt(17)
    for (int j = 0; j < 16; ++j) {
        float acc = q * Wq0[j];
        for (int t = 0; t < 16; ++t) acc += charge_embed[sp * 16 + t] * Wq0[(t + 1) * 16 + j];
        wqbuf[(size_t)i * 16 + j] = (_Float16)(acc * is17);
    }
}

// ---------------- stage 6: van der Waals edge sum ----------------
__global__ void k6_vdw(const float* __restrict__ vectors, const int* __restrict__ senders,
                       const int* __restrict__ receivers, const float* __restrict__ sig_n,
                       const float* __restrict__ epn_n, float* __restrict__ scalars)
{
    int e = blockIdx.x * blockDim.x + threadIdx.x;
    float val = 0.f;
    if (e < E_EDGES) {
        float vx = vectors[e * 3 + 0], vy = vectors[e * 3 + 1], vz = vectors[e * 3 + 2];
        float r = sqrtf(vx * vx + vy * vy + vz * vz);
        float u6 = r * r * r; u6 *= u6;
        float env = 1.f - 28.f * u6 + 48.f * u6 * r - 21.f * u6 * r * r;
        env = (r < 1.f) ? env : 0.f;
        int s = senders[e], t = receivers[e];
        float sij = 0.5f * (sig_n[s] + sig_n[t]);
        float eij = sqrtf(epn_n[s] * epn_n[t]);
        float s1 = sij * frcp(r);
        float s2 = s1 * s1;
        float s6 = s2 * s2 * s2;
        val = 0.5f * 4.f * eij * (s6 * s6 - s6) * env;
    }
    for (int off = 16; off > 0; off >>= 1) val += __shfl_down(val, off, 32);
    if ((threadIdx.x & 31) == 0) atomAddF(&scalars[4], val);
}

// ---------------- stage 7: final edge MLP (K=144) + output head ----------------
__global__ void __launch_bounds__(64) k7_edge_mlp3(
    const float* __restrict__ vectors, const int* __restrict__ senders,
    const _Float16* __restrict__ x2buf, const _Float16* __restrict__ wqbuf,
    const _Float16* __restrict__ pWx20, const _Float16* __restrict__ pWx21,
    const _Float16* __restrict__ pWx22,
    const float* __restrict__ Wout, float* __restrict__ scalars)
{
    __shared__ WS3 ws[2];
    const int wv = threadIdx.x >> 5, lane = threadIdx.x & 31;
    const int half = lane >> 4, ln = lane & 15;
    WS3& W = ws[wv];
    const int tile = blockIdx.x * 2 + wv;
    if (tile * 16 >= E_EDGES) return;
    const int e0 = tile * 16;

    if (half == 0) {
        int e = e0 + ln;
        float vx = vectors[e * 3 + 0], vy = vectors[e * 3 + 1], vz = vectors[e * 3 + 2];
        float r = sqrtf(vx * vx + vy * vy + vz * vz);
        float u6 = r * r * r; u6 *= u6;
        float env = 1.f - 28.f * u6 + 48.f * u6 * r - 21.f * u6 * r * r;
        W.env[ln] = (r < 1.f) ? env : 0.f;
        W.snd[ln] = senders[e];
    }
    {   // A = [x2 (128), wq[sender] (16), 0-pad(16)]
        int e = e0 + ln;
        const _Float16* x2 = x2buf + (size_t)e * HDIM;
        const _Float16* wq = wqbuf + (size_t)W.snd[ln] * 16;
        int k0 = half * 80;
        for (int kk = 0; kk < 80; ++kk) {
            int k = k0 + kk;
            _Float16 v = (k < 128) ? x2[k] : (k < 144 ? wq[k - 128] : (_Float16)0.f);
            W.A0[a_idx(ln, k)] = v;
        }
    }

    wave_gemm16(W.A0, pWx20, 5, 8, true,  nullptr, W.A1, nullptr, nullptr, 0, nullptr, 0, lane);
    wave_gemm16(W.A1, pWx21, 4, 8, true,  nullptr, W.A0, nullptr, nullptr, 0, nullptr, 0, lane);
    wave_gemm16(W.A0, pWx22, 4, 8, false, W.env,   W.A1, nullptr, nullptr, 0, nullptr, 0, lane);

    // out = x3 @ Wout / sqrt(128), summed over all edges
    float val = 0.f;
    if (half == 0) {
        for (int k = 0; k < HDIM; ++k)
            val += (float)W.A1[a_idx(ln, k)] * Wout[k];
        val *= 0.08838834764831845f;  // 1/sqrt(128)
    }
    for (int off = 16; off > 0; off >>= 1) val += __shfl_down(val, off, 32);
    if (lane == 0) atomAddF(&scalars[5], val);
}

// ---------------- stage 8: finalize ----------------
__global__ void k8_final(const float* __restrict__ scalars, float* __restrict__ out)
{
    if (threadIdx.x == 0 && blockIdx.x == 0)
        out[0] = scalars[5] + scalars[2] + scalars[3] + scalars[4];
}

// ======================= host launcher =======================
extern "C" void kernel_launch(void* const* d_in, const int* in_sizes, int n_in,
                              void* d_out, int out_size, void* d_ws, size_t ws_size,
                              hipStream_t stream)
{
    (void)in_sizes; (void)n_in; (void)out_size; (void)ws_size;
    const float* vectors      = (const float*)d_in[0];
    const int*   senders      = (const int*)d_in[1];
    const int*   receivers    = (const int*)d_in[2];
    const int*   species      = (const int*)d_in[3];
    const float* embed        = (const float*)d_in[4];
    const float* pe           = (const float*)d_in[5];
    const float* W2b0         = (const float*)d_in[6];
    const float* W2b1         = (const float*)d_in[7];
    const float* W2b2         = (const float*)d_in[8];
    const float* Wv           = (const float*)d_in[9];
    const float* Wl_w         = (const float*)d_in[10];
    const float* Wl_x0        = (const float*)d_in[11];
    const float* Wl_x1        = (const float*)d_in[12];
    const float* Wl_x2        = (const float*)d_in[13];
    const float* Wchi0        = (const float*)d_in[14];
    const float* Wchi1        = (const float*)d_in[15];
    const float* Wsig0        = (const float*)d_in[16];
    const float* Wsig1        = (const float*)d_in[17];
    const float* Weps0        = (const float*)d_in[18];
    const float* Weps1        = (const float*)d_in[19];
    const float* charge_embed = (const float*)d_in[20];
    const float* Wq0          = (const float*)d_in[21];
    const float* Wx20         = (const float*)d_in[22];
    const float* Wx21         = (const float*)d_in[23];
    const float* Wx22         = (const float*)d_in[24];
    const float* Wout         = (const float*)d_in[25];
    const float* radius       = (const float*)d_in[26];
    const float* hardness     = (const float*)d_in[27];
    const float* varepsilon   = (const float*)d_in[28];

    char* base = (char*)d_ws;
    size_t cur = 0;
    auto alloc = [&](size_t bytes) -> void* {
        void* p = base + cur;
        cur += (bytes + 255) & ~(size_t)255;
        return p;
    };

    // packed weights (f16, B-operand layout)
    _Float16* pW2b0  = (_Float16*)alloc((size_t)96  * 128 * 2);
    _Float16* pW2b1  = (_Float16*)alloc((size_t)128 * 128 * 2);
    _Float16* pW2b2  = (_Float16*)alloc((size_t)128 * 128 * 2);
    _Float16* pWv    = (_Float16*)alloc((size_t)128 * 16  * 2);
    _Float16* pWl_w  = (_Float16*)alloc((size_t)128 * 16  * 2);
    _Float16* pWl_x0 = (_Float16*)alloc((size_t)160 * 128 * 2);
    _Float16* pWl_x1 = (_Float16*)alloc((size_t)128 * 128 * 2);
    _Float16* pWl_x2 = (_Float16*)alloc((size_t)128 * 128 * 2);
    _Float16* pWchi0 = (_Float16*)alloc((size_t)128 * 16  * 2);
    _Float16* pWsig0 = (_Float16*)alloc((size_t)128 * 32  * 2);
    _Float16* pWeps0 = (_Float16*)alloc((size_t)128 * 32  * 2);
    _Float16* pWx20  = (_Float16*)alloc((size_t)160 * 128 * 2);
    _Float16* pWx21  = (_Float16*)alloc((size_t)128 * 128 * 2);
    _Float16* pWx22  = (_Float16*)alloc((size_t)128 * 128 * 2);

    // intermediates
    _Float16* x1buf   = (_Float16*)alloc((size_t)E_EDGES * HDIM * 2);
    _Float16* x2buf   = (_Float16*)alloc((size_t)E_EDGES * HDIM * 2);
    float*    Ybuf    = (float*)alloc((size_t)E_EDGES * 16 * 4);
    float*    vbuf    = (float*)alloc((size_t)E_EDGES * 16 * 4);
    _Float16* scalbuf = (_Float16*)alloc((size_t)E_EDGES * 16 * 2);
    float*    wYacc   = (float*)alloc((size_t)N_ATOMS * 256 * 4);
    _Float16* wqbuf   = (_Float16*)alloc((size_t)N_ATOMS * 16 * 2);
    float*    chis_acc = (float*)alloc((size_t)N_ATOMS * 4);
    float*    sig_acc  = (float*)alloc((size_t)N_ATOMS * 4);
    float*    eps_acc  = (float*)alloc((size_t)N_ATOMS * 4);
    float*    Jbuf     = (float*)alloc((size_t)N_ATOMS * 4);
    float*    sig_n    = (float*)alloc((size_t)N_ATOMS * 4);
    float*    epn_n    = (float*)alloc((size_t)N_ATOMS * 4);
    float*    scalars  = (float*)alloc(8 * 4);

    // zero accumulators (graph-capturable)
    hipMemsetAsync(wYacc,    0, (size_t)N_ATOMS * 256 * 4, stream);
    hipMemsetAsync(chis_acc, 0, (size_t)N_ATOMS * 4, stream);
    hipMemsetAsync(sig_acc,  0, (size_t)N_ATOMS * 4, stream);
    hipMemsetAsync(eps_acc,  0, (size_t)N_ATOMS * 4, stream);
    hipMemsetAsync(scalars,  0, 8 * 4, stream);

    auto pack = [&](const float* W, _Float16* dst, int K, int Kpad, int Nout) {
        float scale = 1.f / sqrtf((float)K);
        int total = Kpad * Nout;
        pack_w<<<dim3((total + 255) / 256), dim3(256), 0, stream>>>(W, dst, K, Kpad, Nout, scale);
    };
    pack(W2b0,  pW2b0,  72,  96,  128);
    pack(W2b1,  pW2b1,  128, 128, 128);
    pack(W2b2,  pW2b2,  128, 128, 128);
    pack(Wv,    pWv,    128, 128, 16);
    pack(Wl_w,  pWl_w,  128, 128, 16);
    pack(Wl_x0, pWl_x0, 144, 160, 128);
    pack(Wl_x1, pWl_x1, 128, 128, 128);
    pack(Wl_x2, pWl_x2, 128, 128, 128);
    pack(Wchi0, pWchi0, 128, 128, 16);
    pack(Wsig0, pWsig0, 128, 128, 32);
    pack(Weps0, pWeps0, 128, 128, 32);
    pack(Wx20,  pWx20,  144, 160, 128);
    pack(Wx21,  pWx21,  128, 128, 128);
    pack(Wx22,  pWx22,  128, 128, 128);

    const int wmma_blocks = E_EDGES / 32;   // 2 waves/block, 16 edges/wave

    k1_edge_mlp<<<dim3(wmma_blocks), dim3(64), 0, stream>>>(
        vectors, senders, receivers, species, embed,
        pW2b0, pW2b1, pW2b2, pWv, pWl_w,
        x1buf, Ybuf, vbuf, wYacc);

    k2_scal<<<dim3((E_EDGES * 16 + 255) / 256), dim3(256), 0, stream>>>(
        senders, wYacc, Ybuf, vbuf, varepsilon, scalbuf);

    k3_edge_mlp2<<<dim3(wmma_blocks), dim3(64), 0, stream>>>(
        vectors, senders, x1buf, scalbuf,
        pWl_x0, pWl_x1, pWl_x2, pWchi0, pWsig0, pWeps0,
        Wchi1, Wsig1, Weps1,
        x2buf, chis_acc, sig_acc, eps_acc);

    k4_node<<<dim3((N_ATOMS + 255) / 256), dim3(256), 0, stream>>>(
        species, radius, hardness, pe, chis_acc, sig_acc, eps_acc,
        Jbuf, sig_n, epn_n, scalars);

    k5_node_q<<<dim3((N_ATOMS + 255) / 256), dim3(256), 0, stream>>>(
        species, chis_acc, Jbuf, charge_embed, Wq0, scalars, wqbuf);

    k6_vdw<<<dim3((E_EDGES + 255) / 256), dim3(256), 0, stream>>>(
        vectors, senders, receivers, sig_n, epn_n, scalars);

    k7_edge_mlp3<<<dim3(wmma_blocks), dim3(64), 0, stream>>>(
        vectors, senders, x2buf, wqbuf, pWx20, pWx21, pWx22, Wout, scalars);

    k8_final<<<dim3(1), dim3(1), 0, stream>>>(scalars, (float*)d_out);
}
